// SetTransformerEncoder_85298050498769
// MI455X (gfx1250) — matrix operand order repaired and checked by reading, not compiled, not run on hardware
//
#include <hip/hip_runtime.h>

// ---------------------------------------------------------------------------
// CDNA5 (gfx1250) Set Transformer encoder: all GEMM/attention contractions on
// v_wmma_f32_16x16x32_bf16 (wave32), f32 accumulation, fused LN epilogues.
// Weights are pre-transposed to [n][k] so every WMMA B-operand is a straight
// 128-bit copy; global->LDS staging uses the gfx1250 async path
// (global_load_async_to_lds_b128 / s_wait_asynccnt) when available.
// ---------------------------------------------------------------------------

typedef __attribute__((ext_vector_type(16))) __bf16 bv16;
typedef __attribute__((ext_vector_type(8)))  float  fv8;
typedef unsigned int u32;
typedef int v4i __attribute__((vector_size(4 * sizeof(int))));

union AFrag { bv16 v; __bf16 e[16]; uint4 q[2]; };
union CFrag { fv8  v; float  e[8];  };

#define WMMA_BF16(A, B, C) \
  __builtin_amdgcn_wmma_f32_16x16x32_bf16(false, (A), false, (B), (short)0, (C), false, false)

#if defined(__AMDGCN__) && __has_builtin(__builtin_amdgcn_global_load_async_to_lds_b128) && \
    __has_builtin(__builtin_amdgcn_s_wait_asynccnt)
#define USE_ASYNC_LDS 1
#else
#define USE_ASYNC_LDS 0
#endif

#if USE_ASYNC_LDS
__device__ __forceinline__ void stage_b128(const __bf16* g, __bf16* l) {
  __builtin_amdgcn_global_load_async_to_lds_b128(
      (__attribute__((address_space(1))) v4i*)g,
      (__attribute__((address_space(3))) v4i*)l, 0, 0);
}
__device__ __forceinline__ void stage_wait() { __builtin_amdgcn_s_wait_asynccnt(0); }
#else
__device__ __forceinline__ void stage_b128(const __bf16* g, __bf16* l) {
  *(uint4*)l = *(const uint4*)g;
}
__device__ __forceinline__ void stage_wait() {}
#endif

__device__ __forceinline__ void cvt8_store(__bf16* dst, const float* src) {
  float4 a = *(const float4*)src;
  float4 b = *(const float4*)(src + 4);
  __bf16 t[8] = {(__bf16)a.x, (__bf16)a.y, (__bf16)a.z, (__bf16)a.w,
                 (__bf16)b.x, (__bf16)b.y, (__bf16)b.z, (__bf16)b.w};
  *(uint4*)dst = *(const uint4*)t;
}

// ------------------ f32 [k][n] -> bf16 transposed [n][k] -------------------
__global__ __launch_bounds__(256) void cvt_bf16_T(const float* __restrict__ in,
                                                  __bf16* __restrict__ out) {
  __shared__ float tile[32][33];
  const int k0 = blockIdx.x * 32, n0 = blockIdx.y * 32, t = threadIdx.x;
  const int r = t >> 3, c4 = (t & 7) * 4;
  const float4 v = *(const float4*)(in + (size_t)(k0 + r) * 256 + n0 + c4);
  tile[r][c4] = v.x; tile[r][c4 + 1] = v.y; tile[r][c4 + 2] = v.z; tile[r][c4 + 3] = v.w;
  __syncthreads();
  __bf16 o[4] = {(__bf16)tile[c4][r], (__bf16)tile[c4 + 1][r],
                 (__bf16)tile[c4 + 2][r], (__bf16)tile[c4 + 3][r]};
  *(uint2*)(out + (size_t)(n0 + r) * 256 + k0 + c4) = *(const uint2*)o;
}

// --------------------------------- GEMM ------------------------------------
// C[M,256] = epilogue(A[M,256] @ W[256,256] + bias);  WT is [n][k] bf16.
// mode 0: bias only.  mode 1: y = LN(resid + relu(acc + bias)) with g/bl.
__global__ __launch_bounds__(256) void gemm256(
    const float* __restrict__ Af, const __bf16* __restrict__ Ab,
    const __bf16* __restrict__ WT, const float* __restrict__ bias,
    const float* __restrict__ resid, const float* __restrict__ g,
    const float* __restrict__ bl, float* __restrict__ of,
    __bf16* __restrict__ ob, int M, int mode) {
  __shared__ __bf16 As[64][40];    // [m][k], padded rows: 20-bank stride
  __shared__ __bf16 BsT[256][40];  // [n][k], padded
  __shared__ float  partial[64][4][2];
  __shared__ float  stats[64][2];

  const int t = threadIdx.x;
  const int wave = t >> 5, lane = t & 31, lr = lane & 15, hf = lane >> 4;
  const int wm = wave >> 2, wn = wave & 3;
  const int m0 = blockIdx.x * 64;

  CFrag acc[2][4];
#pragma unroll
  for (int mt = 0; mt < 2; ++mt)
#pragma unroll
    for (int nt = 0; nt < 4; ++nt)
#pragma unroll
      for (int r = 0; r < 8; ++r) acc[mt][nt].e[r] = 0.f;

  const int ar = t >> 2, ac = (t & 3) * 8;  // A tile: 64 rows x 32 cols, 8/thread
  const int gr = m0 + ar;
  const bool aok = gr < M;
  if (!aok) { uint4 z = {0u, 0u, 0u, 0u}; *(uint4*)&As[ar][ac] = z; }

  for (int kt = 0; kt < 8; ++kt) {
    const int k0 = kt * 32;
    if (aok) {  // A tile (vectorized, bounds hoisted)
      if (Af) {
        cvt8_store(&As[ar][ac], Af + (size_t)gr * 256 + k0 + ac);
        if (kt < 7) __builtin_prefetch(Af + (size_t)gr * 256 + k0 + 32 + ac, 0, 1);
      } else {
        stage_b128(Ab + (size_t)gr * 256 + k0 + ac, &As[ar][ac]);
      }
    }
    {  // B tile: thread t copies WT row t, k-slice [k0, k0+32) — 4x b128
      const __bf16* src = WT + (size_t)t * 256 + k0;
#pragma unroll
      for (int j = 0; j < 4; ++j) stage_b128(src + j * 8, &BsT[t][j * 8]);
    }
    stage_wait();
    __syncthreads();

    AFrag afr[2];
#pragma unroll
    for (int mt = 0; mt < 2; ++mt) {
      const int m = wm * 32 + mt * 16 + lr;
      const int kb = hf * 8;
      afr[mt].q[0] = *(const uint4*)&As[m][kb];
      afr[mt].q[1] = *(const uint4*)&As[m][kb + 16];
    }
    AFrag bfr[4];
#pragma unroll
    for (int nt = 0; nt < 4; ++nt) {
      const int n = wn * 64 + nt * 16 + lr;
      const int kb = hf * 8;
      bfr[nt].q[0] = *(const uint4*)&BsT[n][kb];
      bfr[nt].q[1] = *(const uint4*)&BsT[n][kb + 16];
    }
#pragma unroll
    for (int mt = 0; mt < 2; ++mt)
#pragma unroll
      for (int nt = 0; nt < 4; ++nt)
        acc[mt][nt].v = WMMA_BF16(afr[mt].v, bfr[nt].v, acc[mt][nt].v);
    __syncthreads();
  }

  // epilogue transform: bias (+ relu + residual in FFN mode)
#pragma unroll
  for (int mt = 0; mt < 2; ++mt)
#pragma unroll
    for (int nt = 0; nt < 4; ++nt)
#pragma unroll
      for (int r = 0; r < 8; ++r) {
        const int row = wm * 32 + mt * 16 + hf * 8 + r;
        const int col = wn * 64 + nt * 16 + lr;
        float v = acc[mt][nt].e[r] + bias[col];
        if (mode == 1) {
          const int grow = m0 + row;
          const float rv = (grow < M) ? resid[(size_t)grow * 256 + col] : 0.f;
          v = fmaxf(v, 0.f) + rv;
        }
        acc[mt][nt].e[r] = v;
      }

  if (mode == 1) {  // fused LayerNorm across the 256-wide row
#pragma unroll
    for (int mt = 0; mt < 2; ++mt)
#pragma unroll
      for (int r = 0; r < 8; ++r) {
        float s1 = 0.f, s2 = 0.f;
#pragma unroll
        for (int nt = 0; nt < 4; ++nt) { const float v = acc[mt][nt].e[r]; s1 += v; s2 += v * v; }
#pragma unroll
        for (int msk = 1; msk < 16; msk <<= 1) {
          s1 += __shfl_xor(s1, msk, 32);
          s2 += __shfl_xor(s2, msk, 32);
        }
        if (lr == 0) {
          const int row = wm * 32 + mt * 16 + hf * 8 + r;
          partial[row][wn][0] = s1; partial[row][wn][1] = s2;
        }
      }
    __syncthreads();
    if (t < 64) {
      float s = 0.f, q = 0.f;
      for (int i = 0; i < 4; ++i) { s += partial[t][i][0]; q += partial[t][i][1]; }
      const float mu  = s * (1.f / 256.f);
      const float var = q * (1.f / 256.f) - mu * mu;
      stats[t][0] = mu; stats[t][1] = rsqrtf(var + 1e-5f);
    }
    __syncthreads();
  }

#pragma unroll
  for (int mt = 0; mt < 2; ++mt)
#pragma unroll
    for (int nt = 0; nt < 4; ++nt)
#pragma unroll
      for (int r = 0; r < 8; ++r) {
        const int row = wm * 32 + mt * 16 + hf * 8 + r;
        const int col = wn * 64 + nt * 16 + lr;
        const int grow = m0 + row;
        if (grow >= M) continue;
        float v = acc[mt][nt].e[r];
        if (mode == 1) v = (v - stats[row][0]) * stats[row][1] * g[col] + bl[col];
        if (of) of[(size_t)grow * 256 + col] = v;
        if (ob) ob[(size_t)grow * 256 + col] = (__bf16)v;
      }
}

// -------------------- attention: 32 queries vs 8192 keys -------------------
__global__ __launch_bounds__(256) void attn_qk32(
    const float* __restrict__ Qp, int qp_bstride, const __bf16* __restrict__ K,
    float* __restrict__ logits, float scale) {
  __shared__ __bf16 qs[32][72];
  __shared__ __bf16 ks[256][72];
  const int t = threadIdx.x;
  const int keybase = blockIdx.x * 256, h = blockIdx.y, b = blockIdx.z;

  {  // q: 32x64 f32 -> bf16, 8 contiguous per thread
    const int idx = t * 8, qi = idx >> 6, c = idx & 63;
    cvt8_store(&qs[qi][c], Qp + (size_t)b * qp_bstride + qi * 256 + h * 64 + c);
  }
  {  // keys: one 64-wide row per thread, 8x b128 (async when available)
    const __bf16* src = K + ((size_t)(b * 8192 + keybase + t)) * 256 + h * 64;
#pragma unroll
    for (int j = 0; j < 8; ++j) stage_b128(src + j * 8, &ks[t][j * 8]);
  }
  stage_wait();
  __syncthreads();

  const int wave = t >> 5, lane = t & 31, lr = lane & 15, hf = lane >> 4;
  CFrag acc[2][2];
#pragma unroll
  for (int mt = 0; mt < 2; ++mt)
#pragma unroll
    for (int nt = 0; nt < 2; ++nt)
#pragma unroll
      for (int r = 0; r < 8; ++r) acc[mt][nt].e[r] = 0.f;

#pragma unroll
  for (int kt = 0; kt < 2; ++kt) {
    const int kb0 = kt * 32;
    AFrag afr[2];
#pragma unroll
    for (int mt = 0; mt < 2; ++mt) {
      const int m = mt * 16 + lr, kb = kb0 + hf * 8;
      afr[mt].q[0] = *(const uint4*)&qs[m][kb];
      afr[mt].q[1] = *(const uint4*)&qs[m][kb + 16];
    }
    AFrag bfr[2];
#pragma unroll
    for (int nt = 0; nt < 2; ++nt) {
      const int key = wave * 32 + nt * 16 + lr, kb = kb0 + hf * 8;
      bfr[nt].q[0] = *(const uint4*)&ks[key][kb];
      bfr[nt].q[1] = *(const uint4*)&ks[key][kb + 16];
    }
#pragma unroll
    for (int mt = 0; mt < 2; ++mt)
#pragma unroll
      for (int nt = 0; nt < 2; ++nt)
        acc[mt][nt].v = WMMA_BF16(afr[mt].v, bfr[nt].v, acc[mt][nt].v);
  }

  const size_t rbase = ((size_t)(b * 4 + h)) * 32;
#pragma unroll
  for (int mt = 0; mt < 2; ++mt)
#pragma unroll
    for (int nt = 0; nt < 2; ++nt)
#pragma unroll
      for (int r = 0; r < 8; ++r) {
        const int row = mt * 16 + hf * 8 + r;
        const int key = keybase + wave * 32 + nt * 16 + lr;
        logits[(rbase + row) * 8192 + key] = acc[mt][nt].e[r] * scale;
      }
}

// ------------------- softmax over 8192, write bf16 in place ----------------
__global__ __launch_bounds__(256) void softmax_rows(
    const float* __restrict__ logits, __bf16* __restrict__ P, int len) {
  __shared__ float buf[8192];
  __shared__ float red[256];
  const int r = blockIdx.x, t = threadIdx.x;
  const float* src = logits + (size_t)r * len;
  float mx = -1e30f;
  for (int i = t; i < len; i += 256) { const float v = src[i]; buf[i] = v; mx = fmaxf(mx, v); }
  red[t] = mx; __syncthreads();
  for (int s = 128; s > 0; s >>= 1) { if (t < s) red[t] = fmaxf(red[t], red[t + s]); __syncthreads(); }
  mx = red[0]; __syncthreads();
  float sum = 0.f;
  for (int i = t; i < len; i += 256) { const float e = __expf(buf[i] - mx); buf[i] = e; sum += e; }
  red[t] = sum; __syncthreads();
  for (int s = 128; s > 0; s >>= 1) { if (t < s) red[t] += red[t + s]; __syncthreads(); }
  const float inv = 1.f / red[0];
  __bf16* dst = P + (size_t)r * len;
  for (int i = t; i < len; i += 256) dst[i] = (__bf16)(buf[i] * inv);
}

// ------------- O[b,32,h*64:..] = Qp + P[32,8192] @ V[8192,64] --------------
__global__ __launch_bounds__(256) void attn_pv32(
    const __bf16* __restrict__ P, const __bf16* __restrict__ V,
    const float* __restrict__ Qp, int qp_bstride, float* __restrict__ O) {
  __shared__ float  accs[32][64];
  __shared__ __bf16 vstg[8][64][40];  // per-wave transposed V tile [n][k]
  const int t = threadIdx.x;
  const int h = blockIdx.x, b = blockIdx.y;
  const int wave = t >> 5, lane = t & 31, lr = lane & 15, hf = lane >> 4;
#pragma unroll
  for (int j = 0; j < 2; ++j) {
    const int idx = t * 2 + j;  // 512 float4 slots
    ((float4*)&accs[0][0])[idx] = float4{0.f, 0.f, 0.f, 0.f};
  }
  __syncthreads();

  CFrag acc[2][4];
#pragma unroll
  for (int mt = 0; mt < 2; ++mt)
#pragma unroll
    for (int nt = 0; nt < 4; ++nt)
#pragma unroll
      for (int r = 0; r < 8; ++r) acc[mt][nt].e[r] = 0.f;

  const size_t pb = ((size_t)(b * 4 + h)) * 32 * 8192;
  for (int kc = 0; kc < 32; ++kc) {  // each wave owns 1024 keys
    const int kb = wave * 1024 + kc * 32;
    // stage V[kb..kb+31][h*64..+63] transposed: lane = (2 k-rows) x (32-col half)
    {
      const int krow = kb + 2 * lr;
      const int c0 = hf * 32;
      const u32* r0 = (const u32*)(V + ((size_t)(b * 8192 + krow)) * 256 + h * 64 + c0);
      const u32* r1 = (const u32*)(V + ((size_t)(b * 8192 + krow + 1)) * 256 + h * 64 + c0);
#pragma unroll
      for (int j = 0; j < 16; ++j) {
        const u32 u0 = r0[j], u1 = r1[j];
        *(u32*)&vstg[wave][c0 + 2 * j][2 * lr]     = (u0 & 0xffffu) | (u1 << 16);
        *(u32*)&vstg[wave][c0 + 2 * j + 1][2 * lr] = (u0 >> 16) | (u1 & 0xffff0000u);
      }
    }
    AFrag afr[2];
#pragma unroll
    for (int mt = 0; mt < 2; ++mt) {
      const int m = mt * 16 + lr; const int klo = kb + hf * 8;
      afr[mt].q[0] = *(const uint4*)(P + pb + (size_t)m * 8192 + klo);
      afr[mt].q[1] = *(const uint4*)(P + pb + (size_t)m * 8192 + klo + 16);
    }
    AFrag bfr[4];
#pragma unroll
    for (int nt = 0; nt < 4; ++nt) {
      const int n = nt * 16 + lr, kb2 = hf * 8;
      bfr[nt].q[0] = *(const uint4*)&vstg[wave][n][kb2];
      bfr[nt].q[1] = *(const uint4*)&vstg[wave][n][kb2 + 16];
    }
#pragma unroll
    for (int mt = 0; mt < 2; ++mt)
#pragma unroll
      for (int nt = 0; nt < 4; ++nt)
        acc[mt][nt].v = WMMA_BF16(afr[mt].v, bfr[nt].v, acc[mt][nt].v);
  }
  // cross-wave K reduction through LDS float atomics (ds_add_f32)
#pragma unroll
  for (int mt = 0; mt < 2; ++mt)
#pragma unroll
    for (int nt = 0; nt < 4; ++nt)
#pragma unroll
      for (int r = 0; r < 8; ++r)
        atomicAdd(&accs[mt * 16 + hf * 8 + r][nt * 16 + lr], acc[mt][nt].e[r]);
  __syncthreads();

#pragma unroll
  for (int j = 0; j < 8; ++j) {
    const int idx = t * 8 + j, qi = idx >> 6, c = idx & 63;
    const float v = accs[qi][c] + Qp[(size_t)b * qp_bstride + qi * 256 + h * 64 + c];
    O[((size_t)(b * 32 + qi)) * 256 + h * 64 + c] = v;
  }
}

// ---- fused attention for mab1: 8192 queries vs 32 keys, in-place on QO ----
__global__ __launch_bounds__(256) void attn_small_kv(
    float* __restrict__ QO, const __bf16* __restrict__ K,
    const __bf16* __restrict__ V, float scale) {
  __shared__ __bf16 qs[64][72];
  __shared__ __bf16 ks[32][72];
  __shared__ __bf16 vstg[64][40];  // transposed V: [n][k]
  __shared__ __bf16 ps[64][40];
  __shared__ float  ls[64][33];
  const int t = threadIdx.x;
  const int rb = blockIdx.x, h = blockIdx.y, b = blockIdx.z;
  const int wave = t >> 5, lane = t & 31, lr = lane & 15, hf = lane >> 4;
  const size_t qbase = ((size_t)b * 8192 + rb * 64) * 256 + h * 64;

  {  // 64x64 q block f32 -> bf16, 16 contiguous per thread
    const int idx = t * 16, r = idx >> 6, c = idx & 63;
    cvt8_store(&qs[r][c],     QO + qbase + (size_t)r * 256 + c);
    cvt8_store(&qs[r][c + 8], QO + qbase + (size_t)r * 256 + c + 8);
  }
  {  // K row-major copy (b128)
    const int kj = t >> 3, c0 = (t & 7) * 8;
    stage_b128(K + ((size_t)b * 32 + kj) * 256 + h * 64 + c0, &ks[kj][c0]);
  }
  {  // V transposed via 2x2 halfword interleave: 2 k-rows x 4 cols per thread
    const int kjp = t >> 4, c0 = (t & 15) * 4;
    const u32* r0 = (const u32*)(V + ((size_t)b * 32 + 2 * kjp) * 256 + h * 64 + c0);
    const u32* r1 = (const u32*)(V + ((size_t)b * 32 + 2 * kjp + 1) * 256 + h * 64 + c0);
#pragma unroll
    for (int j = 0; j < 2; ++j) {
      const u32 u0 = r0[j], u1 = r1[j];
      *(u32*)&vstg[c0 + 2 * j][2 * kjp]     = (u0 & 0xffffu) | (u1 << 16);
      *(u32*)&vstg[c0 + 2 * j + 1][2 * kjp] = (u0 >> 16) | (u1 & 0xffff0000u);
    }
  }
  stage_wait();
  __syncthreads();

  {  // S = q @ k^T * scale   (4 m-tiles x 2 n-tiles over 8 waves)
    const int mt = wave >> 1, nt = wave & 1;
    CFrag acc;
#pragma unroll
    for (int r = 0; r < 8; ++r) acc.e[r] = 0.f;
#pragma unroll
    for (int kt = 0; kt < 2; ++kt) {
      const int kb0 = kt * 32;
      AFrag afr;
      { const int m = mt * 16 + lr, kb = kb0 + hf * 8;
        afr.q[0] = *(const uint4*)&qs[m][kb];
        afr.q[1] = *(const uint4*)&qs[m][kb + 16]; }
      AFrag bfr;
      { const int key = nt * 16 + lr, kb = kb0 + hf * 8;
        bfr.q[0] = *(const uint4*)&ks[key][kb];
        bfr.q[1] = *(const uint4*)&ks[key][kb + 16]; }
      acc.v = WMMA_BF16(afr.v, bfr.v, acc.v);
    }
#pragma unroll
    for (int r = 0; r < 8; ++r) ls[mt * 16 + hf * 8 + r][nt * 16 + lr] = acc.e[r] * scale;
  }
  __syncthreads();

  if (t < 64) {  // softmax over 32 keys per query row
    float mx = -1e30f;
#pragma unroll
    for (int j = 0; j < 32; ++j) mx = fmaxf(mx, ls[t][j]);
    float sum = 0.f;
#pragma unroll
    for (int j = 0; j < 32; ++j) { const float e = __expf(ls[t][j] - mx); ls[t][j] = e; sum += e; }
    const float inv = 1.f / sum;
#pragma unroll
    for (int j = 0; j < 32; ++j) ps[t][j] = (__bf16)(ls[t][j] * inv);
  }
  __syncthreads();

  {  // O = P @ V (single K=32 step), residual-add in place
    const int mt = wave >> 1;
#pragma unroll
    for (int ntl = 0; ntl < 2; ++ntl) {
      const int nt = (wave & 1) * 2 + ntl;
      CFrag oc;
#pragma unroll
      for (int r = 0; r < 8; ++r) oc.e[r] = 0.f;
      AFrag afr;
      { const int m = mt * 16 + lr, kb = hf * 8;
        afr.q[0] = *(const uint4*)&ps[m][kb];
        afr.q[1] = *(const uint4*)&ps[m][kb + 16]; }
      AFrag bfr;
      { const int n = nt * 16 + lr, kb2 = hf * 8;
        bfr.q[0] = *(const uint4*)&vstg[n][kb2];
        bfr.q[1] = *(const uint4*)&vstg[n][kb2 + 16]; }
      oc.v = WMMA_BF16(afr.v, bfr.v, oc.v);
#pragma unroll
      for (int r = 0; r < 8; ++r) {
        const int row = mt * 16 + hf * 8 + r, col = nt * 16 + lr;
        const size_t gi = qbase + (size_t)row * 256 + col;
        QO[gi] = QO[gi] + oc.e[r];
      }
    }
  }
}

// ------------------------- LayerNorm (in place) ----------------------------
__global__ __launch_bounds__(256) void layernorm_rows(
    float* __restrict__ x, const float* __restrict__ g, const float* __restrict__ b, int M) {
  __shared__ float red[256];
  __shared__ float red2[256];
  const int row = blockIdx.x, t = threadIdx.x;
  const float v = x[(size_t)row * 256 + t];
  red[t] = v; red2[t] = v * v; __syncthreads();
  for (int s = 128; s > 0; s >>= 1) {
    if (t < s) { red[t] += red[t + s]; red2[t] += red2[t + s]; }
    __syncthreads();
  }
  const float mu  = red[0] * (1.f / 256.f);
  const float var = red2[0] * (1.f / 256.f) - mu * mu;
  const float rstd = rsqrtf(var + 1e-5f);
  x[(size_t)row * 256 + t] = (v - mu) * rstd * g[t] + b[t];
}

// ---------------- final: out[b] = mean_seeds(latents) @ Wout + bout --------
__global__ __launch_bounds__(256) void final_out(
    const float* __restrict__ lat, const float* __restrict__ Wout,
    const float* __restrict__ bout, float* __restrict__ out) {
  __shared__ float mean[256];
  const int b = blockIdx.x, t = threadIdx.x;
  float s = 0.f;
#pragma unroll
  for (int i = 0; i < 32; ++i) s += lat[((size_t)(b * 32 + i)) * 256 + t];
  mean[t] = s * (1.f / 32.f);
  __syncthreads();
  float acc = bout[t];
  for (int k = 0; k < 256; ++k) acc += mean[k] * Wout[(size_t)k * 256 + t];
  out[(size_t)b * 256 + t] = acc;
}

// ---------------------------------------------------------------------------
extern "C" void kernel_launch(void* const* d_in, const int* in_sizes, int n_in,
                              void* d_out, int out_size, void* d_ws, size_t ws_size,
                              hipStream_t stream) {
  (void)in_sizes; (void)n_in; (void)out_size; (void)ws_size;
  const int B = 8, N = 8192, NB = B * N;
  const float scale = 0.0625f;  // 1/sqrt(256)

  auto F = [&](int i) { return (const float*)d_in[i]; };
  // d_in layout: 0 x_patches, 1 Win, 2 b_in, 3 I, 4..15 mab0, 16..27 mab1,
  //              28 S, 29..40 mabp, 41 Wout, 42 bout
  const int M0 = 4, M1 = 16, MP = 29;
  // mab offsets: Wq+0 bq+1 Wk+2 bk+3 Wv+4 bv+5 Wo+6 bo+7 g0+8 b0+9 g1+10 b1+11

  char* ws = (char*)d_ws;
  size_t off = 0;
  auto alloc = [&](size_t bytes) -> char* {
    size_t o = (off + 255) & ~(size_t)255; off = o + bytes; return ws + o;
  };

  __bf16* wWin = (__bf16*)alloc(65536 * 2);
  __bf16* w0q = (__bf16*)alloc(65536 * 2); __bf16* w0k = (__bf16*)alloc(65536 * 2);
  __bf16* w0v = (__bf16*)alloc(65536 * 2); __bf16* w0o = (__bf16*)alloc(65536 * 2);
  __bf16* w1q = (__bf16*)alloc(65536 * 2); __bf16* w1k = (__bf16*)alloc(65536 * 2);
  __bf16* w1v = (__bf16*)alloc(65536 * 2); __bf16* w1o = (__bf16*)alloc(65536 * 2);
  __bf16* wpq = (__bf16*)alloc(65536 * 2); __bf16* wpk = (__bf16*)alloc(65536 * 2);
  __bf16* wpv = (__bf16*)alloc(65536 * 2); __bf16* wpo = (__bf16*)alloc(65536 * 2);

  __bf16* xb = (__bf16*)alloc((size_t)NB * 256 * 2);          // x (and later x2) bf16
  __bf16* Kp = (__bf16*)alloc((size_t)NB * 256 * 2);          // Kp0 / Kp(PMA)
  __bf16* Vp = (__bf16*)alloc((size_t)NB * 256 * 2);          // Vp0 / Vp(PMA)
  float*  logits = (float*)alloc((size_t)1024 * 8192 * 4);    // logits; P bf16 overlays
  __bf16* P = (__bf16*)logits;
  float*  Qbig = (float*)alloc((size_t)NB * 256 * 4);         // Qp1 / O1 / ln in place
  float*  Qp0f = (float*)alloc(32 * 256 * 4);
  float*  O0   = (float*)alloc((size_t)B * 32 * 256 * 4);
  __bf16* Htb  = (__bf16*)alloc((size_t)B * 32 * 256 * 2);
  __bf16* K1b  = (__bf16*)alloc((size_t)B * 32 * 256 * 2);
  __bf16* V1b  = (__bf16*)alloc((size_t)B * 32 * 256 * 2);
  float*  Qpsf = (float*)alloc(32 * 256 * 4);
  float*  Osml = (float*)alloc((size_t)B * 32 * 256 * 4);
  float*  lat  = (float*)alloc((size_t)B * 32 * 256 * 4);

  auto convT = [&](const float* in, __bf16* out) {  // [k][n] f32 -> [n][k] bf16
    cvt_bf16_T<<<dim3(8, 8), 256, 0, stream>>>(in, out);
  };
  convT(F(1), wWin);
  convT(F(M0 + 0), w0q); convT(F(M0 + 2), w0k); convT(F(M0 + 4), w0v); convT(F(M0 + 6), w0o);
  convT(F(M1 + 0), w1q); convT(F(M1 + 2), w1k); convT(F(M1 + 4), w1v); convT(F(M1 + 6), w1o);
  convT(F(MP + 0), wpq); convT(F(MP + 2), wpk); convT(F(MP + 4), wpv); convT(F(MP + 6), wpo);

  auto gemm = [&](const float* Af, const __bf16* Ab, const __bf16* WT, const float* bias,
                  const float* resid, const float* g, const float* bl,
                  float* of, __bf16* ob, int M, int mode) {
    gemm256<<<dim3((M + 63) / 64), 256, 0, stream>>>(Af, Ab, WT, bias, resid, g, bl, of, ob, M, mode);
  };

  // input projection: xb = bf16(x_patches @ Win + b_in)
  gemm(F(0), nullptr, wWin, F(2), nullptr, nullptr, nullptr, nullptr, xb, NB, 0);

  // ---- mab0 (ISAB half 1): Q = I (broadcast), K = x ----
  gemm(F(3), nullptr, w0q, F(M0 + 1), nullptr, nullptr, nullptr, Qp0f, nullptr, 32, 0);
  gemm(nullptr, xb, w0k, F(M0 + 3), nullptr, nullptr, nullptr, nullptr, Kp, NB, 0);
  gemm(nullptr, xb, w0v, F(M0 + 5), nullptr, nullptr, nullptr, nullptr, Vp, NB, 0);
  attn_qk32<<<dim3(32, 4, 8), 256, 0, stream>>>(Qp0f, 0, Kp, logits, scale);
  softmax_rows<<<1024, 256, 0, stream>>>(logits, P, 8192);
  attn_pv32<<<dim3(4, 8), 256, 0, stream>>>(P, Vp, Qp0f, 0, O0);
  layernorm_rows<<<B * 32, 256, 0, stream>>>(O0, F(M0 + 8), F(M0 + 9), B * 32);
  gemm(O0, nullptr, w0o, F(M0 + 7), O0, F(M0 + 10), F(M0 + 11), nullptr, Htb, B * 32, 1);

  // ---- mab1 (ISAB half 2): Q = x, K = Ht ----
  gemm(nullptr, xb, w1q, F(M1 + 1), nullptr, nullptr, nullptr, Qbig, nullptr, NB, 0);
  gemm(nullptr, Htb, w1k, F(M1 + 3), nullptr, nullptr, nullptr, nullptr, K1b, B * 32, 0);
  gemm(nullptr, Htb, w1v, F(M1 + 5), nullptr, nullptr, nullptr, nullptr, V1b, B * 32, 0);
  attn_small_kv<<<dim3(128, 4, 8), 256, 0, stream>>>(Qbig, K1b, V1b, scale);
  layernorm_rows<<<NB, 256, 0, stream>>>(Qbig, F(M1 + 8), F(M1 + 9), NB);
  gemm(Qbig, nullptr, w1o, F(M1 + 7), Qbig, F(M1 + 10), F(M1 + 11), nullptr, xb, NB, 1);

  // ---- PMA: Q = S (broadcast), K = x2 ----
  gemm(F(28), nullptr, wpq, F(MP + 1), nullptr, nullptr, nullptr, Qpsf, nullptr, 32, 0);
  gemm(nullptr, xb, wpk, F(MP + 3), nullptr, nullptr, nullptr, nullptr, Kp, NB, 0);
  gemm(nullptr, xb, wpv, F(MP + 5), nullptr, nullptr, nullptr, nullptr, Vp, NB, 0);
  attn_qk32<<<dim3(32, 4, 8), 256, 0, stream>>>(Qpsf, 0, Kp, logits, scale);
  softmax_rows<<<1024, 256, 0, stream>>>(logits, P, 8192);
  attn_pv32<<<dim3(4, 8), 256, 0, stream>>>(P, Vp, Qpsf, 0, Osml);
  layernorm_rows<<<B * 32, 256, 0, stream>>>(Osml, F(MP + 8), F(MP + 9), B * 32);
  gemm(Osml, nullptr, wpo, F(MP + 7), Osml, F(MP + 10), F(MP + 11), lat, nullptr, B * 32, 1);

  // ---- output head ----
  final_out<<<8, 256, 0, stream>>>(lat, F(41), F(42), (float*)d_out);
}
